// AttentionHead_1881195675910
// MI455X (gfx1250) — compile-verified, matrix operand order checked by reading, not run on hardware
//
#include <hip/hip_runtime.h>

#define B_ 8
#define S_ 2048
#define E_ 1024
#define D_ 64

typedef __attribute__((ext_vector_type(16))) __bf16          v16bf;
typedef __attribute__((ext_vector_type(8)))  float           v8f;
typedef __attribute__((ext_vector_type(16))) unsigned short  v16u;
typedef __attribute__((ext_vector_type(8)))  unsigned short  v8u;

// fp32 -> bf16 via native conversion (RNE): lowers to v_cvt_pk_bf16_f32
static __device__ __forceinline__ unsigned short f2bf(float f) {
  __bf16 h = (__bf16)f;
  return __builtin_bit_cast(unsigned short, h);
}

static __device__ __forceinline__ v16bf as_bf(v16u u) {
  return __builtin_bit_cast(v16bf, u);
}

// 16-element bf16 operand fragment from two contiguous 8xu16 chunks
// (chunk0 at p, chunk1 at p+16 elems), matching the 16-bit WMMA layout:
// lane = khalf*16 + idx, VGPR0-3 = K[khalf*8..+7], VGPR4-7 = K[16+khalf*8..+7].
static __device__ __forceinline__ v16bf load_frag_u16(const unsigned short* p) {
  v8u lo = *(const v8u*)(p);
  v8u hi = *(const v8u*)(p + 16);
  v16u t;
#pragma unroll
  for (int i = 0; i < 8; ++i) { t[i] = lo[i]; t[i + 8] = hi[i]; }
  return as_bf(t);
}

// ---------------------------------------------------------------------------
// Kernel 0: convert Wq|Wk|Wv (fp32, 3 x 64 x 1024) to bf16 once (384 KB,
// L2-resident for the whole projection pass).
// ---------------------------------------------------------------------------
__global__ void __launch_bounds__(256)
wconv_kernel(const float* __restrict__ Wq, const float* __restrict__ Wk,
             const float* __restrict__ Wv, unsigned short* __restrict__ Wbf) {
  const int i = blockIdx.x * 256 + threadIdx.x;   // 0 .. 64*1024-1
  Wbf[i]                 = f2bf(Wq[i]);
  Wbf[i + D_ * E_]       = f2bf(Wk[i]);
  Wbf[i + 2 * D_ * E_]   = f2bf(Wv[i]);
}

// ---------------------------------------------------------------------------
// Kernel 1: fused Q/K/V projection.  One wave per (b, 16-row s-tile) computes
// all three 16x64 outputs (12 accumulator tiles); each x element is converted
// to bf16 exactly once.  Q pre-scaled by 1/sqrt(D); V stored transposed.
// ---------------------------------------------------------------------------
__global__ void __launch_bounds__(256)
qkv_proj_kernel(const float* __restrict__ x,
                const unsigned short* __restrict__ Wbf,
                unsigned short* __restrict__ Qb,
                unsigned short* __restrict__ Kb,
                unsigned short* __restrict__ Vtb) {
  const int wave = threadIdx.x >> 5;
  const int lane = threadIdx.x & 31;
  const int col  = lane & 15;   // M for A-frag / N for B-frag / N for C
  const int grp  = lane >> 4;   // K-half for A/B frags, M-half for C

  const int tile = blockIdx.x * 8 + wave;      // b * (S/16) + s-tile
  const int b    = tile / (S_ / 16);
  const int s0   = (tile % (S_ / 16)) * 16;

  v8f acc[12] = {};                            // [m*4 + t], m: 0=Q 1=K 2=V

  const float* xrow = x + ((long)(b * S_) + s0 + col) * E_;

  for (int e0 = 0; e0 < E_; e0 += 32) {
    v16bf a;
    {
      const float* p = xrow + e0 + grp * 8;
#pragma unroll
      for (int i = 0; i < 8; ++i) {
        a[i]     = (__bf16)p[i];
        a[i + 8] = (__bf16)p[i + 16];
      }
    }
#pragma unroll
    for (int m = 0; m < 3; ++m) {
#pragma unroll
      for (int t = 0; t < 4; ++t) {
        const unsigned short* wr =
            Wbf + (long)m * D_ * E_ + (long)(t * 16 + col) * E_ + e0 + grp * 8;
        acc[m * 4 + t] = __builtin_amdgcn_wmma_f32_16x16x32_bf16(
            false, a, false, load_frag_u16(wr), (short)0, acc[m * 4 + t],
            false, false);
      }
    }
  }

  {
    unsigned short* q  = Qb + ((long)(b * S_) + s0) * D_;
    unsigned short* k  = Kb + ((long)(b * S_) + s0) * D_;
    unsigned short* vt = Vtb + (long)b * D_ * S_;
#pragma unroll
    for (int t = 0; t < 4; ++t)
#pragma unroll
      for (int r = 0; r < 8; ++r) {
        const int row = r + 8 * grp;
        q[(long)row * D_ + t * 16 + col]         = f2bf(acc[t][r] * 0.125f);
        k[(long)row * D_ + t * 16 + col]         = f2bf(acc[4 + t][r]);
        vt[(long)(t * 16 + col) * S_ + s0 + row] = f2bf(acc[8 + t][r]);
      }
  }
}

// ---------------------------------------------------------------------------
// Kernel 2: causal flash attention.  One wave owns a 16-query tile and the
// full 16x64 fp32 output accumulator; iterates over 64-key steps (4 S-tiles,
// 16 WMMAs per step) so the per-step softmax fixed cost (shuffle reductions,
// O rescale) is amortized over 2x the elements.  Masking only on the
// (wave-uniform) diagonal-crossing step.
// ---------------------------------------------------------------------------
__global__ void __launch_bounds__(256)
flash_attn_kernel(const unsigned short* __restrict__ Qb,
                  const unsigned short* __restrict__ Kb,
                  const unsigned short* __restrict__ Vtb,
                  float* __restrict__ out) {
  __shared__ unsigned short lds[8][16 * 64];   // per-wave P staging (bf16)

  const int wave = threadIdx.x >> 5;
  const int lane = threadIdx.x & 31;
  const int col  = lane & 15;
  const int grp  = lane >> 4;

  const int g  = blockIdx.x * 8 + wave;        // global q-tile id
  const int b  = g / (S_ / 16);
  const int q0 = (g % (S_ / 16)) * 16;

  // Q A-fragments: 2 K-steps of 32 across D=64 (Q already scaled by 1/sqrt(D))
  const unsigned short* qrow = Qb + ((long)(b * S_) + q0 + col) * D_;
  v16bf aq[2];
#pragma unroll
  for (int kk = 0; kk < 2; ++kk)
    aq[kk] = load_frag_u16(qrow + kk * 32 + grp * 8);

  v8f o[4] = {};
  float mstat[8], lstat[8];
#pragma unroll
  for (int r = 0; r < 8; ++r) { mstat[r] = -1e30f; lstat[r] = 0.f; }

  unsigned short* pl = lds[wave];

  for (int kv0 = 0; kv0 < q0 + 16; kv0 += 64) {
    // ---- S = Q * K^T : four 16x16 N-tiles, two K-steps each ----
    v8f sacc[4] = {};
#pragma unroll
    for (int u = 0; u < 4; ++u) {
#pragma unroll
      for (int kk = 0; kk < 2; ++kk) {
        const unsigned short* kr =
            Kb + ((long)(b * S_) + kv0 + u * 16 + col) * D_ + kk * 32 + grp * 8;
        sacc[u] = __builtin_amdgcn_wmma_f32_16x16x32_bf16(
            false, aq[kk], false, load_frag_u16(kr), (short)0, sacc[u], false, false);
      }
    }

    // ---- causal mask (only on diagonal-crossing step; wave-uniform) ----
    const bool needmask = (kv0 + 63 > q0);
    if (needmask) {
#pragma unroll
      for (int r = 0; r < 8; ++r) {
        const int qidx = q0 + r + 8 * grp;
#pragma unroll
        for (int u = 0; u < 4; ++u) {
          const int key = kv0 + u * 16 + col;
          if (key > qidx) sacc[u][r] = -1e30f;
        }
      }
    }

    // ---- row max (half-wave shuffle reduce, once per 64 keys) ----
    float rm[8];
#pragma unroll
    for (int r = 0; r < 8; ++r) {
      float m0 = fmaxf(fmaxf(sacc[0][r], sacc[1][r]),
                       fmaxf(sacc[2][r], sacc[3][r]));
#pragma unroll
      for (int off = 1; off < 16; off <<= 1)
        m0 = fmaxf(m0, __shfl_xor(m0, off, 16));
      rm[r] = m0;
    }

    // ---- online softmax update ----
#pragma unroll
    for (int r = 0; r < 8; ++r) {
      const float mn    = fmaxf(mstat[r], rm[r]);
      const float alpha = __expf(mstat[r] - mn);
      mstat[r] = mn;
      float rs = 0.f;
#pragma unroll
      for (int u = 0; u < 4; ++u) {
        const float pv = __expf(sacc[u][r] - mn);
        sacc[u][r] = pv;
        rs += pv;
      }
#pragma unroll
      for (int off = 1; off < 16; off <<= 1)
        rs += __shfl_xor(rs, off, 16);
      lstat[r] = lstat[r] * alpha + rs;
#pragma unroll
      for (int t = 0; t < 4; ++t) o[t][r] *= alpha;
    }

    // ---- C-layout P -> LDS (row-major bf16 16x64) -> A-layout fragments ----
#pragma unroll
    for (int u = 0; u < 4; ++u)
#pragma unroll
      for (int r = 0; r < 8; ++r)
        pl[(r + 8 * grp) * 64 + u * 16 + col] = f2bf(sacc[u][r]);
    // intra-wave DS ops are in-order: no barrier required
    v16bf pa[2];
#pragma unroll
    for (int kk = 0; kk < 2; ++kk)
      pa[kk] = load_frag_u16(pl + col * 64 + kk * 32 + grp * 8);

    // ---- O += P * V  (V transposed: contiguous key chunks per d-row) ----
#pragma unroll
    for (int t = 0; t < 4; ++t) {
#pragma unroll
      for (int kk = 0; kk < 2; ++kk) {
        const unsigned short* vr =
            Vtb + ((long)b * D_ + t * 16 + col) * S_ + kv0 + kk * 32 + grp * 8;
        o[t] = __builtin_amdgcn_wmma_f32_16x16x32_bf16(
            false, pa[kk], false, load_frag_u16(vr), (short)0, o[t], false, false);
      }
    }
  }

  // ---- epilogue: O / l ----
  float* orow = out + ((long)(b * S_) + q0) * D_;
#pragma unroll
  for (int r = 0; r < 8; ++r) {
    const float inv = 1.f / lstat[r];
#pragma unroll
    for (int t = 0; t < 4; ++t)
      orow[(long)(r + 8 * grp) * D_ + t * 16 + col] = o[t][r] * inv;
  }
}

// ---------------------------------------------------------------------------
extern "C" void kernel_launch(void* const* d_in, const int* in_sizes, int n_in,
                              void* d_out, int out_size, void* d_ws, size_t ws_size,
                              hipStream_t stream) {
  const float* x  = (const float*)d_in[0];
  const float* Wq = (const float*)d_in[1];
  const float* Wk = (const float*)d_in[2];
  const float* Wv = (const float*)d_in[3];
  // d_in[4] = attention_mask: causal, computed analytically in-kernel.

  unsigned short* Qb  = (unsigned short*)d_ws;                 // 2 MB
  unsigned short* Kb  = Qb + (size_t)B_ * S_ * D_;             // 2 MB
  unsigned short* Vtb = Kb + (size_t)B_ * S_ * D_;             // 2 MB (transposed)
  unsigned short* Wbf = Vtb + (size_t)B_ * S_ * D_;            // 384 KB (Q|K|V)

  wconv_kernel<<<(D_ * E_) / 256, 256, 0, stream>>>(Wq, Wk, Wv, Wbf);
  qkv_proj_kernel<<<B_ * (S_ / 16) / 8, 256, 0, stream>>>(x, Wbf, Qb, Kb, Vtb);
  flash_attn_kernel<<<B_ * (S_ / 16) / 8, 256, 0, stream>>>(Qb, Kb, Vtb,
                                                            (float*)d_out);
}
